// ReliabilityWeightedFusion_60198261621272
// MI455X (gfx1250) — matrix operand order repaired and checked by reading, not compiled, not run on hardware
//
#include <hip/hip_runtime.h>

#define S_  14
#define B_  16384
#define L_  30
#define CH_ 64
#define F_  32
#define RH_ 16

typedef __attribute__((ext_vector_type(16))) __bf16 v16bf;
typedef __attribute__((ext_vector_type(8)))  __bf16 v8bf;
typedef __attribute__((ext_vector_type(2)))  __bf16 v2bf;
typedef __attribute__((ext_vector_type(8)))  float  v8f;

// per-batch-row stride in bf16 elements: 32 lp slots * 64 ch + 8 pad
// (pad makes lane stride 1028 dwords == 4 mod 64 banks -> conflict-free b128)
#define H1_BSTRIDE 2056

// ---------------------------------------------------------------------------
// Kernel 1: per (sensor, 16-batch tile): conv1 (VALU) -> LDS bf16,
// conv2 as WMMA GEMM (M=16 batch, N=16 f per wave, K=192 im2col) with a
// rolling A-fragment window (only 2 new ds_load_b128 pairs per position),
// ReLU + mean over L, then fc1/fc2 reliability logits.
// Writes feats (S,B,F) and logits (S,B) to workspace.
// ---------------------------------------------------------------------------
__global__ __launch_bounds__(64)
void rwf_encoder_kernel(const float* __restrict__ x_g,
                        const float* __restrict__ w1_g,
                        const float* __restrict__ b1_g,
                        const float* __restrict__ w2_g,
                        const float* __restrict__ b2_g,
                        const float* __restrict__ fc1w_g,
                        const float* __restrict__ fc1b_g,
                        const float* __restrict__ fc2w_g,
                        const float* __restrict__ fc2b_g,
                        float* __restrict__ feats_ws,
                        float* __restrict__ logits_ws)
{
    __shared__ float xs[16][32];                                    // padded L
    __shared__ __attribute__((aligned(16))) __bf16 h1[16 * H1_BSTRIDE];
    __shared__ float feat_s[16][F_];

    const int tid   = threadIdx.x;
    const int btile = blockIdx.x;       // 0..1023
    const int s     = blockIdx.y;       // 0..13
    const int b0    = btile * 16;

    // ---- stage x tile into LDS, zero-padded at lp=0 and lp=31 ----
    if (tid < 32) xs[tid >> 1][(tid & 1) * 31] = 0.0f;
    for (int i = tid; i < 16 * L_; i += 64) {
        int b = i / L_, l = i % L_;
        xs[b][l + 1] = x_g[(size_t)(s * B_ + b0 + b) * L_ + l];
    }
    __syncthreads();

    // ---- conv1 + ReLU -> h1 (bf16) layout [b][lp][c] ----
    {
        const int pr = tid & 31;          // channel pair 0..31
        const int bh = tid >> 5;          // batch half
        const int c0 = pr * 2;
        const float w00 = w1_g[(s * CH_ + c0) * 3 + 0];
        const float w01 = w1_g[(s * CH_ + c0) * 3 + 1];
        const float w02 = w1_g[(s * CH_ + c0) * 3 + 2];
        const float w10 = w1_g[(s * CH_ + c0 + 1) * 3 + 0];
        const float w11 = w1_g[(s * CH_ + c0 + 1) * 3 + 1];
        const float w12 = w1_g[(s * CH_ + c0 + 1) * 3 + 2];
        const float bb0 = b1_g[s * CH_ + c0];
        const float bb1 = b1_g[s * CH_ + c0 + 1];

        v2bf zz; zz.x = (__bf16)0.0f; zz.y = (__bf16)0.0f;
        for (int b = bh * 8; b < bh * 8 + 8; ++b) {
            __bf16* row = &h1[b * H1_BSTRIDE];
            *(v2bf*)&row[0 * 64 + c0]  = zz;      // lp = 0  pad
            *(v2bf*)&row[31 * 64 + c0] = zz;      // lp = 31 pad
            for (int l = 0; l < L_; ++l) {
                float xm = xs[b][l], xc = xs[b][l + 1], xp = xs[b][l + 2];
                float v0 = bb0 + w00 * xm + w01 * xc + w02 * xp;
                float v1 = bb1 + w10 * xm + w11 * xc + w12 * xp;
                v0 = v0 > 0.0f ? v0 : 0.0f;
                v1 = v1 > 0.0f ? v1 : 0.0f;
                v2bf p; p.x = (__bf16)v0; p.y = (__bf16)v1;
                *(v2bf*)&row[(l + 1) * 64 + c0] = p;
            }
        }
    }
    __syncthreads();

    // ---- conv2 as WMMA: wave w handles f in [w*16, w*16+16) ----
    const int lane   = tid & 31;
    const int fbase  = (tid >> 5) * 16;
    const int nlo    = lane & 15;
    const bool hiH   = (lane & 16) != 0;
    const int fcol   = fbase + nlo;

    // B fragments: K x N = 32 x 16, lanes 0-15 hold K {0..7,16..23} (pairs per VGPR),
    // lanes 16-31 hold K {8..15,24..31}. K = tap*64 + c (tap-major im2col).
    // bfrag index = tap*2 + chalf (chalf: c 0-31 vs 32-63).
    v16bf bfrag[6];
    {
        const int kl0 = hiH ? 8 : 0;
        #pragma unroll
        for (int kc = 0; kc < 6; ++kc) {
            const int tap = kc >> 1;
            const int cbs = (kc & 1) * 32;
            #pragma unroll
            for (int j = 0; j < 16; ++j) {
                int c = cbs + kl0 + j + ((j >> 3) << 3);
                bfrag[kc][j] =
                    (__bf16)w2_g[((size_t)(s * F_ + fcol) * CH_ + c) * 3 + tap];
            }
        }
    }
    const float bias2 = b2_g[s * F_ + fcol];

    // A fragment addressing: lanes 0-15 -> K {0..7,16..23}, 16-31 -> {8..15,24..31}
    const int m    = lane & 15;
    const int koff = hiH ? 8 : 0;
    const __bf16* hbase = &h1[m * H1_BSTRIDE + koff];

    auto loadA = [&](int lp, int chalf) -> v16bf {
        const __bf16* p = hbase + lp * 64 + chalf * 32;
        v8bf lo = *(const v8bf*)p;            // K kl..kl+7   (ds_load_b128)
        v8bf hi = *(const v8bf*)(p + 16);     // K kl+16..+23 (ds_load_b128)
        return __builtin_shufflevector(lo, hi,
                   0,1,2,3,4,5,6,7,8,9,10,11,12,13,14,15);
    };

    float fsum[8];
    #pragma unroll
    for (int r = 0; r < 8; ++r) fsum[r] = 0.0f;

    // Rolling window of A fragments over lp (position l reads lp = l, l+1, l+2).
    v16bf a00 = loadA(0, 0), a01 = loadA(0, 1);   // lp = l
    v16bf a10 = loadA(1, 0), a11 = loadA(1, 1);   // lp = l+1

    #pragma unroll 3
    for (int l = 0; l < L_; ++l) {
        v16bf a20 = loadA(l + 2, 0), a21 = loadA(l + 2, 1);   // lp = l+2

        v8f acc = {0.f, 0.f, 0.f, 0.f, 0.f, 0.f, 0.f, 0.f};
        acc = __builtin_amdgcn_wmma_f32_16x16x32_bf16(false, a00, false, bfrag[0], (short)0, acc, false, false);
        acc = __builtin_amdgcn_wmma_f32_16x16x32_bf16(false, a01, false, bfrag[1], (short)0, acc, false, false);
        acc = __builtin_amdgcn_wmma_f32_16x16x32_bf16(false, a10, false, bfrag[2], (short)0, acc, false, false);
        acc = __builtin_amdgcn_wmma_f32_16x16x32_bf16(false, a11, false, bfrag[3], (short)0, acc, false, false);
        acc = __builtin_amdgcn_wmma_f32_16x16x32_bf16(false, a20, false, bfrag[4], (short)0, acc, false, false);
        acc = __builtin_amdgcn_wmma_f32_16x16x32_bf16(false, a21, false, bfrag[5], (short)0, acc, false, false);

        #pragma unroll
        for (int r = 0; r < 8; ++r) {
            float v = acc[r] + bias2;             // bias + ReLU + mean-accum
            fsum[r] += (v > 0.0f) ? v : 0.0f;
        }

        a00 = a10; a01 = a11;                     // rotate window (renamed by
        a10 = a20; a11 = a21;                     // the x3 unroll)
    }

    // D layout: lane carries N=f column; VGPR r -> M row (r, or r+8 for hi lanes)
    const int brow0 = hiH ? 8 : 0;
    #pragma unroll
    for (int r = 0; r < 8; ++r) {
        float fv = fsum[r] * (1.0f / 30.0f);
        int brow = brow0 + r;
        feat_s[brow][fcol] = fv;
        feats_ws[(size_t)(s * B_ + b0 + brow) * F_ + fcol] = fv;
    }
    __syncthreads();

    // ---- reliability head: one lane per batch row (512 MACs each) ----
    if (tid < 16) {
        const int b = tid;
        float logit = fc2b_g[s];
        for (int j = 0; j < RH_; ++j) {
            float acc = fc1b_g[s * RH_ + j];
            #pragma unroll
            for (int f = 0; f < F_; ++f)
                acc += feat_s[b][f] * fc1w_g[(size_t)(s * RH_ + j) * F_ + f];
            acc = acc > 0.0f ? acc : 0.0f;
            logit += acc * fc2w_g[s * RH_ + j];
        }
        logits_ws[s * B_ + b0 + b] = logit;
    }
}

// ---------------------------------------------------------------------------
// Kernel 2: softmax over S, weighted fusion, predictor MLP, outputs.
// d_out = [ rul (B) | weights (B,S) ]
// ---------------------------------------------------------------------------
__global__ __launch_bounds__(256)
void rwf_head_kernel(const float* __restrict__ feats_ws,
                     const float* __restrict__ logits_ws,
                     const float* __restrict__ p1w, const float* __restrict__ p1b,
                     const float* __restrict__ p2w, const float* __restrict__ p2b,
                     const float* __restrict__ p3w, const float* __restrict__ p3b,
                     float* __restrict__ out)
{
    int b = blockIdx.x * 256 + threadIdx.x;
    if (b >= B_) return;

    float lg[S_];
    float mx = -3.4e38f;
    #pragma unroll
    for (int s = 0; s < S_; ++s) {
        lg[s] = logits_ws[s * B_ + b];
        mx = fmaxf(mx, lg[s]);
    }
    float sum = 0.0f;
    #pragma unroll
    for (int s = 0; s < S_; ++s) { lg[s] = __expf(lg[s] - mx); sum += lg[s]; }
    const float inv = 1.0f / sum;

    float fused[F_];
    #pragma unroll
    for (int f = 0; f < F_; ++f) fused[f] = 0.0f;

    for (int s = 0; s < S_; ++s) {
        float w = lg[s] * inv;
        out[B_ + (size_t)b * S_ + s] = w;                 // weights (B,S,1)
        const float* fp = &feats_ws[(size_t)(s * B_ + b) * F_];
        #pragma unroll
        for (int f = 0; f < F_; ++f) fused[f] += w * fp[f];
    }

    float z1[64];
    for (int i = 0; i < 64; ++i) {
        float a = p1b[i];
        #pragma unroll
        for (int f = 0; f < F_; ++f) a += fused[f] * p1w[i * F_ + f];
        z1[i] = a > 0.0f ? a : 0.0f;
    }
    float z2[32];
    for (int j = 0; j < 32; ++j) {
        float a = p2b[j];
        #pragma unroll
        for (int i = 0; i < 64; ++i) a += z1[i] * p2w[j * 64 + i];
        z2[j] = a > 0.0f ? a : 0.0f;
    }
    float r = p3b[0];
    #pragma unroll
    for (int j = 0; j < 32; ++j) r += z2[j] * p3w[j];
    out[b] = r;                                           // rul (B,1)
}

extern "C" void kernel_launch(void* const* d_in, const int* in_sizes, int n_in,
                              void* d_out, int out_size, void* d_ws, size_t ws_size,
                              hipStream_t stream)
{
    const float* x    = (const float*)d_in[0];
    const float* w1   = (const float*)d_in[1];
    const float* b1   = (const float*)d_in[2];
    const float* w2   = (const float*)d_in[3];
    const float* b2   = (const float*)d_in[4];
    const float* fc1w = (const float*)d_in[5];
    const float* fc1b = (const float*)d_in[6];
    const float* fc2w = (const float*)d_in[7];
    const float* fc2b = (const float*)d_in[8];
    const float* p1w  = (const float*)d_in[9];
    const float* p1b  = (const float*)d_in[10];
    const float* p2w  = (const float*)d_in[11];
    const float* p2b  = (const float*)d_in[12];
    const float* p3w  = (const float*)d_in[13];
    const float* p3b  = (const float*)d_in[14];
    float* out = (float*)d_out;

    float* feats_ws  = (float*)d_ws;                       // S*B*F fp32
    float* logits_ws = feats_ws + (size_t)S_ * B_ * F_;    // S*B fp32

    dim3 g1(B_ / 16, S_), t1(64);
    rwf_encoder_kernel<<<g1, t1, 0, stream>>>(x, w1, b1, w2, b2,
                                              fc1w, fc1b, fc2w, fc2b,
                                              feats_ws, logits_ws);

    rwf_head_kernel<<<dim3((B_ + 255) / 256), dim3(256), 0, stream>>>(
        feats_ws, logits_ws, p1w, p1b, p2w, p2b, p3w, p3b, out);
}
